// TransformerNet_15496242004785
// MI455X (gfx1250) — matrix operand order repaired and checked by reading, not compiled, not run on hardware
//
#include <hip/hip_runtime.h>
#include <math.h>

// ---------------------------------------------------------------------------
// Model constants (match reference)
// ---------------------------------------------------------------------------
#define EDIM   512
#define HIDDIM 256
#define NLAYER 2
#define NCELLS 131072
#define NDONOR 256
#define MROWS  64          // rows of cells per workgroup (4 WMMA M-tiles)
#define MT     (MROWS/16)  // M-tiles per block
#define LN_EPS 1e-5f

// LDS strides padded so row-to-row bank offset == 4 (conflict-free fragment
// loads with ds_load_b128), all 16-byte aligned.
#define XLDF   (EDIM + 4)     // f32 residual stride (words)   516
#define LDH    (EDIM + 8)     // f16 E-wide stride (halves)    520
#define LDH2   (HIDDIM + 8)   // f16 HID-wide stride (halves)  264
#define LDTF   (HIDDIM + 4)   // f32 tanh-buffer stride        260

// f16 weight area layout (in halves), per layer: Wv, Wo, W1, W2
#define OFF_WV  0
#define OFF_WO  (EDIM*EDIM)                 // 262144
#define OFF_W1  (2*EDIM*EDIM)               // 524288
#define OFF_W2  (2*EDIM*EDIM + HIDDIM*EDIM) // 655360
#define PER_LAYER_H (2*EDIM*EDIM + 2*HIDDIM*EDIM)   // 786432
#define OFF_APW1 (NLAYER*PER_LAYER_H)               // 1572864

typedef __attribute__((ext_vector_type(16))) _Float16 v16h;
typedef __attribute__((ext_vector_type(8)))  float    v8f;

union Frag16 { v16h v; float4 q[2]; };

// ---------------------------------------------------------------------------
// WMMA helpers (CDNA5 wave32, 16x16x32 f16 -> f32)
// ---------------------------------------------------------------------------
__device__ __forceinline__ v8f wmma_f16(v16h a, v16h b, v8f c) {
  // (neg_a, A, neg_b, B, c_mod, C, reuse_a, reuse_b)
  return __builtin_amdgcn_wmma_f32_16x16x32_f16(false, a, false, b, (short)0, c,
                                                false, false);
}

// A-matrix fragment (16x32, f16, row-major in LDS, row stride ldk halves).
// ISA layout: lanes 0-15 hold K={kb..kb+7, kb+16..kb+23},
//             lanes 16-31 hold K={kb+8..kb+15, kb+24..kb+31}; M = lane%16.
__device__ __forceinline__ v16h load_a_frag(const _Float16* A, int ldk, int kb) {
  const int lane = threadIdx.x & 31;
  const int m = lane & 15;
  const int b = kb + ((lane >> 4) << 3);
  const _Float16* p = A + m * ldk + b;
  Frag16 f;
  f.q[0] = *(const float4*)(p);        // 8 halves: K = b .. b+7
  f.q[1] = *(const float4*)(p + 16);   // 8 halves: K = b+16 .. b+23
  return f.v;
}

// B-matrix fragment (32x16). Weight stored as (N_out, K_in) row-major, so
// B[k][n] = W[n*ldk + k]. ISA layout: N = lane%16, lanes 0-15 hold K={kb..kb+7, kb+16..kb+23},
// lanes 16-31 hold K={kb+8..kb+15, kb+24..kb+31} (same order as A).
__device__ __forceinline__ v16h load_b_frag(const _Float16* W, int ldk, int n0, int kb) {
  const int lane = threadIdx.x & 31;
  const int n = n0 + (lane & 15);
  const int b = kb + ((lane >> 4) << 3);
  const _Float16* p = W + (size_t)n * ldk + b;
  Frag16 f;
  f.q[0] = *(const float4*)(p);
  f.q[1] = *(const float4*)(p + 16);
  return f.v;
}

// C/D layout: VGPR r, lane l -> M = m0 + r + 8*(l/16), N = n0 + l%16.
// EPI: 0 = bias -> f16 store, 1 = bias+relu -> f16 store,
//      2 = bias + residual add into f32 LDS, 3 = bias+tanh -> f32 store
template <int EPI>
__device__ __forceinline__ void epilogue(v8f c, void* Dst, int ldd, int m0,
                                         int n0, const float* bias) {
  const int lane = threadIdx.x & 31;
  const int n = n0 + (lane & 15);
  const int mb = m0 + ((lane >> 4) << 3);
  const float bb = bias[n];
#pragma unroll
  for (int r = 0; r < 8; ++r) {
    float v = c[r] + bb;
    if (EPI == 0) {
      ((_Float16*)Dst)[(mb + r) * ldd + n] = (_Float16)v;
    } else if (EPI == 1) {
      ((_Float16*)Dst)[(mb + r) * ldd + n] = (_Float16)fmaxf(v, 0.f);
    } else if (EPI == 2) {
      ((float*)Dst)[(mb + r) * ldd + n] += v;
    } else {
      ((float*)Dst)[(mb + r) * ldd + n] = tanhf(v);
    }
  }
}

// Tiled GEMM for one MROWS-row block: C(MROWS x NT*16) = A(MROWS x KDIM)*W^T.
// 8 waves; each wave owns NT/16 column-tile pairs and reuses each B-fragment
// pair across all MT row tiles (4x reduction in weight traffic vs MT=1).
template <int NT, int KDIM, int EPI>
__device__ __forceinline__ void gemm_tile(const _Float16* A, int lda,
                                          const _Float16* W, int ldw,
                                          const float* bias, void* Dst, int ldd,
                                          int wave) {
  for (int i = 0; i < NT / 16; ++i) {
    const int n0a = (2 * wave + 16 * i) * 16;
    const int n0b = n0a + 16;
    v8f acc[MT][2];
#pragma unroll
    for (int m = 0; m < MT; ++m) { acc[m][0] = (v8f){}; acc[m][1] = (v8f){}; }
    for (int kb = 0; kb < KDIM; kb += 32) {
      v16h b0 = load_b_frag(W, ldw, n0a, kb);
      v16h b1 = load_b_frag(W, ldw, n0b, kb);
#pragma unroll
      for (int m = 0; m < MT; ++m) {
        v16h a = load_a_frag(A + m * 16 * lda, lda, kb);
        acc[m][0] = wmma_f16(a, b0, acc[m][0]);
        acc[m][1] = wmma_f16(a, b1, acc[m][1]);
      }
    }
#pragma unroll
    for (int m = 0; m < MT; ++m) {
      epilogue<EPI>(acc[m][0], Dst, ldd, m * 16, n0a, bias);
      epilogue<EPI>(acc[m][1], Dst, ldd, m * 16, n0b, bias);
    }
  }
}

// LayerNorm of MROWSx512 f32 tile -> f16 tile. 4 lanes per row; xor-shuffles
// with mask < 4 stay inside each 4-lane row group of a wave32.
__device__ __forceinline__ void layernorm_f16(const float* xf, _Float16* h,
                                              const float* g, const float* b) {
  const int r = threadIdx.x >> 2;
  const int j = threadIdx.x & 3;
  float s = 0.f, sq = 0.f;
  for (int c = j; c < EDIM; c += 4) {
    float v = xf[r * XLDF + c];
    s += v;
    sq += v * v;
  }
#pragma unroll
  for (int m = 2; m >= 1; m >>= 1) {
    s += __shfl_xor(s, m, 32);
    sq += __shfl_xor(sq, m, 32);
  }
  const float mu = s * (1.f / EDIM);
  const float rs = rsqrtf(sq * (1.f / EDIM) - mu * mu + LN_EPS);
  for (int c = j; c < EDIM; c += 4) {
    float v = (xf[r * XLDF + c] - mu) * rs * g[c] + b[c];
    h[r * LDH + c] = (_Float16)v;
  }
}

// ---------------------------------------------------------------------------
// Kernel 0: f32 -> f16 weight conversion
// ---------------------------------------------------------------------------
__global__ void cvt_f32_to_f16_kernel(const float* __restrict__ src,
                                      _Float16* __restrict__ dst, int n) {
  int i = blockIdx.x * 256 + threadIdx.x;
  if (i < n) dst[i] = (_Float16)src[i];
}

// ---------------------------------------------------------------------------
// Kernel 1: fused 2-layer encoder + attention-pool scorer.
// NOTE: softmax over an axis of size 1 == 1, so attn == V exactly; Q/K are
// mathematically dead and skipped.
// ---------------------------------------------------------------------------
struct EncParams {
  const float* x;
  const _Float16* w16;
  const float* ln1_g[NLAYER];
  const float* ln1_b[NLAYER];
  const float* bv[NLAYER];
  const float* bo[NLAYER];
  const float* ln2_g[NLAYER];
  const float* ln2_b[NLAYER];
  const float* b1[NLAYER];
  const float* b2[NLAYER];
  const float* ap_b1;
  const float* ap_W2;
  const float* ap_b2;
  float* cf;     // [NCELLS, EDIM] final features
  float* score;  // [NCELLS] attention-pool logit
};

__launch_bounds__(256) __global__ void encoder_fused_kernel(EncParams p) {
  // 259 KB static LDS (CDNA5 allows up to 320 KB per workgroup)
  __shared__ __align__(16) float xf[MROWS * XLDF];      // f32 residual stream
  __shared__ __align__(16) _Float16 hA[MROWS * LDH];    // f16 GEMM input ping
  __shared__ __align__(16) _Float16 hB[MROWS * LDH];    // f16 pong / f32 tanh
  const int tid = threadIdx.x;
  const int wave = tid >> 5;
  const size_t r0 = (size_t)blockIdx.x * MROWS;

  // Load MROWSx512 f32 tile (vectorized, row-wise to honor LDS padding)
  {
    const float4* xin = (const float4*)(p.x + r0 * EDIM);
    for (int idx = tid; idx < MROWS * (EDIM / 4); idx += 256) {
      const int row = idx >> 7;       // EDIM/4 = 128 float4 per row
      const int c4 = idx & 127;
      *(float4*)(xf + row * XLDF + c4 * 4) = xin[idx];
    }
  }
  __syncthreads();

  for (int L = 0; L < NLAYER; ++L) {
    const _Float16* Wv = p.w16 + (size_t)L * PER_LAYER_H + OFF_WV;
    const _Float16* Wo = p.w16 + (size_t)L * PER_LAYER_H + OFF_WO;
    const _Float16* W1 = p.w16 + (size_t)L * PER_LAYER_H + OFF_W1;
    const _Float16* W2 = p.w16 + (size_t)L * PER_LAYER_H + OFF_W2;

    layernorm_f16(xf, hA, p.ln1_g[L], p.ln1_b[L]);
    __syncthreads();
    // v = LN1(x) @ Wv^T + bv   (== attention output; softmax over len-1 == 1)
    gemm_tile<32, EDIM, 0>(hA, LDH, Wv, EDIM, p.bv[L], hB, LDH, wave);
    __syncthreads();
    // x += v @ Wo^T + bo
    gemm_tile<32, EDIM, 2>(hB, LDH, Wo, EDIM, p.bo[L], xf, XLDF, wave);
    __syncthreads();
    layernorm_f16(xf, hA, p.ln2_g[L], p.ln2_b[L]);
    __syncthreads();
    // t = relu(LN2(x) @ W1^T + b1)  (MROWS x 256, f16)
    gemm_tile<16, EDIM, 1>(hA, LDH, W1, EDIM, p.b1[L], hB, LDH2, wave);
    __syncthreads();
    // x += t @ W2^T + b2
    gemm_tile<32, HIDDIM, 2>(hB, LDH2, W2, HIDDIM, p.b2[L], xf, XLDF, wave);
    __syncthreads();
  }

  // Emit cf (f32) and convert to f16 for the attention-pool GEMM.
  {
    float4* cfo = (float4*)(p.cf + r0 * EDIM);
    for (int idx = tid; idx < MROWS * (EDIM / 4); idx += 256) {
      const int row = idx >> 7;
      const int c4 = idx & 127;
      cfo[idx] = *(const float4*)(xf + row * XLDF + c4 * 4);
    }
    for (int idx = tid; idx < MROWS * EDIM; idx += 256) {
      const int row = idx >> 9;
      const int c = idx & 511;
      hA[row * LDH + c] = (_Float16)xf[row * XLDF + c];
    }
  }
  __syncthreads();

  // t = tanh(cf @ apW1^T + ap_b1)  -> MROWS x 256 f32 (aliases hB exactly)
  float* tf = (float*)hB;
  gemm_tile<16, EDIM, 3>(hA, LDH, p.w16 + OFF_APW1, EDIM, p.ap_b1, tf, LDTF,
                         wave);
  __syncthreads();

  // score[row] = t[row] . apW2 + ap_b2   (4 lanes per row)
  {
    const int r = tid >> 2, j = tid & 3;
    float acc = 0.f;
    for (int c = j; c < HIDDIM; c += 4) acc += tf[r * LDTF + c] * p.ap_W2[c];
#pragma unroll
    for (int m = 2; m >= 1; m >>= 1) acc += __shfl_xor(acc, m, 32);
    if (j == 0) p.score[r0 + r] = acc + p.ap_b2[0];
  }
}

// ---------------------------------------------------------------------------
// Kernel 2: per-donor softmax pooling (donor_ids are sorted -> contiguous).
// ---------------------------------------------------------------------------
__launch_bounds__(256) __global__ void pool_kernel(const int* __restrict__ donor,
                                                   const float* __restrict__ s,
                                                   const float* __restrict__ cf,
                                                   float* __restrict__ pooled) {
  const int d = blockIdx.x;
  __shared__ float red[256];
  __shared__ int seg[2];
  if (threadIdx.x == 0) {
    int lo = 0, hi = NCELLS;
    while (lo < hi) { int mid = (lo + hi) >> 1; if (donor[mid] < d) lo = mid + 1; else hi = mid; }
    seg[0] = lo;
    lo = 0; hi = NCELLS;
    while (lo < hi) { int mid = (lo + hi) >> 1; if (donor[mid] < d + 1) lo = mid + 1; else hi = mid; }
    seg[1] = lo;
  }
  __syncthreads();
  const int start = seg[0], end = seg[1];

  float m = -INFINITY;
  for (int i = start + threadIdx.x; i < end; i += 256) m = fmaxf(m, s[i]);
  red[threadIdx.x] = m;
  __syncthreads();
  for (int step = 128; step; step >>= 1) {
    if (threadIdx.x < step) red[threadIdx.x] = fmaxf(red[threadIdx.x], red[threadIdx.x + step]);
    __syncthreads();
  }
  m = red[0];
  __syncthreads();

  float ds = 0.f;
  for (int i = start + threadIdx.x; i < end; i += 256) ds += expf(s[i] - m);
  red[threadIdx.x] = ds;
  __syncthreads();
  for (int step = 128; step; step >>= 1) {
    if (threadIdx.x < step) red[threadIdx.x] += red[threadIdx.x + step];
    __syncthreads();
  }
  const float inv = (red[0] > 0.f) ? 1.f / red[0] : 0.f;

  float a0 = 0.f, a1 = 0.f;
  for (int i = start; i < end; ++i) {
    const float w = expf(s[i] - m) * inv;
    const float* row = cf + (size_t)i * EDIM;
    a0 += w * row[threadIdx.x];
    a1 += w * row[threadIdx.x + 256];
  }
  pooled[d * EDIM + threadIdx.x] = a0;
  pooled[d * EDIM + threadIdx.x + 256] = a1;
}

// ---------------------------------------------------------------------------
// Kernel 3: five head GEMVs. out = concat(thal[256x6], braak[256x7],
// cerad[256x4], adnc[256x4], reg[256x4]) flattened row-major per head.
// ---------------------------------------------------------------------------
__launch_bounds__(64) __global__ void heads_kernel(
    const float* __restrict__ pooled, const float* Wt, const float* bt,
    const float* Wb, const float* bb, const float* Wc, const float* bc,
    const float* Wa, const float* ba, const float* Wr, const float* br,
    float* __restrict__ out) {
  __shared__ float row[EDIM];
  const int d = blockIdx.x;
  for (int i = threadIdx.x; i < EDIM; i += 64) row[i] = pooled[d * EDIM + i];
  __syncthreads();
  const int t = threadIdx.x;
  if (t >= 25) return;
  const float* W; const float* b; int off, j, dim;
  if (t < 6)       { W = Wt; b = bt; off = 0;    j = t;      dim = 6; }
  else if (t < 13) { W = Wb; b = bb; off = 1536; j = t - 6;  dim = 7; }
  else if (t < 17) { W = Wc; b = bc; off = 3328; j = t - 13; dim = 4; }
  else if (t < 21) { W = Wa; b = ba; off = 4352; j = t - 17; dim = 4; }
  else             { W = Wr; b = br; off = 5376; j = t - 21; dim = 4; }
  float acc = b[j];
  for (int k = 0; k < EDIM; ++k) acc += row[k] * W[j * EDIM + k];
  out[off + d * dim + j] = acc;
}

// ---------------------------------------------------------------------------
// Host launcher.
// Input flattening assumption (setup_inputs() insertion order, recursive):
//   0: x, 1: donor_ids,
//   layer L (base = 2 + 16*L): ln1_g, ln1_b, Wq, bq, Wk, bk, Wv, bv, Wo, bo,
//                              ln2_g, ln2_b, W1, b1, W2, b2
//   34: ap_W1, 35: ap_b1, 36: ap_W2, 37: ap_b2,
//   38..47: thal_W/b, braak_W/b, cerad_W/b, adnc_W/b, reg_W/b
// ---------------------------------------------------------------------------
extern "C" void kernel_launch(void* const* d_in, const int* in_sizes, int n_in,
                              void* d_out, int out_size, void* d_ws,
                              size_t ws_size, hipStream_t stream) {
  (void)in_sizes; (void)n_in; (void)out_size; (void)ws_size;
  const float* x = (const float*)d_in[0];
  const int* donor = (const int*)d_in[1];
  auto F = [&](int i) { return (const float*)d_in[i]; };

  // Workspace carve: cf (256MB f32) | score | pooled | f16 weights (~3.25MB)
  float* cf = (float*)d_ws;
  float* sc = cf + (size_t)NCELLS * EDIM;
  float* pooled = sc + NCELLS;
  _Float16* w16 = (_Float16*)(pooled + (size_t)NDONOR * EDIM);

  // Convert live weights to f16 (Wq/Wk are mathematically dead -> skipped).
  struct Cv { int idx; size_t off; int n; };
  const Cv cvs[9] = {
      {2 + 6,  0 * PER_LAYER_H + OFF_WV, EDIM * EDIM},
      {2 + 8,  0 * PER_LAYER_H + OFF_WO, EDIM * EDIM},
      {2 + 12, 0 * PER_LAYER_H + OFF_W1, HIDDIM * EDIM},
      {2 + 14, 0 * PER_LAYER_H + OFF_W2, EDIM * HIDDIM},
      {18 + 6, 1 * PER_LAYER_H + OFF_WV, EDIM * EDIM},
      {18 + 8, 1 * PER_LAYER_H + OFF_WO, EDIM * EDIM},
      {18 + 12, 1 * PER_LAYER_H + OFF_W1, HIDDIM * EDIM},
      {18 + 14, 1 * PER_LAYER_H + OFF_W2, EDIM * HIDDIM},
      {34, OFF_APW1, HIDDIM * EDIM},
  };
  for (int c = 0; c < 9; ++c)
    cvt_f32_to_f16_kernel<<<(cvs[c].n + 255) / 256, 256, 0, stream>>>(
        F(cvs[c].idx), w16 + cvs[c].off, cvs[c].n);

  EncParams ep;
  ep.x = x; ep.w16 = w16; ep.cf = cf; ep.score = sc;
  for (int L = 0; L < NLAYER; ++L) {
    const int b = 2 + 16 * L;
    ep.ln1_g[L] = F(b + 0);  ep.ln1_b[L] = F(b + 1);
    ep.bv[L]    = F(b + 7);  ep.bo[L]    = F(b + 9);
    ep.ln2_g[L] = F(b + 10); ep.ln2_b[L] = F(b + 11);
    ep.b1[L]    = F(b + 13); ep.b2[L]    = F(b + 15);
  }
  ep.ap_b1 = F(35); ep.ap_W2 = F(36); ep.ap_b2 = F(37);

  encoder_fused_kernel<<<NCELLS / MROWS, 256, 0, stream>>>(ep);
  pool_kernel<<<NDONOR, 256, 0, stream>>>(donor, sc, cf, pooled);
  heads_kernel<<<NDONOR, 64, 0, stream>>>(pooled, F(38), F(39), F(40), F(41),
                                          F(42), F(43), F(44), F(45), F(46),
                                          F(47), (float*)d_out);
}
